// NAFBlockHyper_87763361727184
// MI455X (gfx1250) — matrix operand order repaired and checked by reading, not compiled, not run on hardware
//
#include <hip/hip_runtime.h>

// ---------------------------------------------------------------------------
// NAFNet block, MI455X (gfx1250), wave32 + v_wmma_f32_16x16x32_f16.
// All 1x1 convs run as 16x16x32 f16 WMMAs with fp32 accumulation.
// Memory-bound target ~670MB HBM traffic @ 23.3 TB/s.
// ---------------------------------------------------------------------------

typedef _Float16 half16 __attribute__((ext_vector_type(16)));
typedef float    float8 __attribute__((ext_vector_type(8)));

#define NB 8
#define NC 64
#define NDW 128
#define NHW 65536      // 256*256
#define TILES_PER_B 512  // 65536 / 128 pixels

union F16Frag { half16 v; uint4 q[2]; };

// A-matrix fragment (16x32 f16, row-major weights [rows][64] in global).
// Per ISA 16-bit A layout: lane<16 holds M=lane, K{khalf*8..+7} in v0-3 and
// K{16+khalf*8..+7} in v4-7 with khalf=lane>>4.
static __device__ __forceinline__ half16
load_a_frag(const _Float16* __restrict__ wrow, int lane, int kt) {
  int khalf = (lane >> 4) & 1;
  const _Float16* p = wrow + kt * 32 + khalf * 8;
  F16Frag f;
  f.q[0] = *reinterpret_cast<const uint4*>(p);       // K = base .. base+7
  f.q[1] = *reinterpret_cast<const uint4*>(p + 16);  // K = base+16 .. base+23
  return f.v;
}

// B-matrix fragment (32x16 f16). Lane = column N (lane&15); khalf=lane>>4 holds
// K=kt*32+khalf*8..+7 in v0-3 and K=kt*32+16+khalf*8..+7 in v4-7 (matches A).
static __device__ __forceinline__ half16
load_b_frag(const _Float16* actrow, int lane, int kt) {
  int kb = ((lane >> 4) & 1) * 8 + kt * 32;
  const _Float16* p = actrow + kb;
  F16Frag f;
  f.q[0] = *reinterpret_cast<const uint4*>(p);
  f.q[1] = *reinterpret_cast<const uint4*>(p + 16);
  return f.v;
}

// C init from per-output-channel bias (broadcast across columns).
static __device__ __forceinline__ float8
bias_acc(const float* __restrict__ bias, int mbase) {
  float8 c;
#pragma unroll
  for (int r = 0; r < 8; ++r) c[r] = bias[mbase + r];
  return c;
}

static __device__ __forceinline__ float8 wmma32(half16 a, half16 b, float8 c) {
  return __builtin_amdgcn_wmma_f32_16x16x32_f16(false, a, false, b,
                                                (short)0, c, false, false);
}

// ---------------------------------------------------------------------------
// Kernel 1: per-(b, out-channel) Frobenius weight normalization (+cast f16).
// ---------------------------------------------------------------------------
__global__ void k_prep(const float* __restrict__ w, _Float16* __restrict__ dh,
                       float* __restrict__ df, int rowLen) {
  int row = blockIdx.x;
  int lane = threadIdx.x;
  const float* src = w + (size_t)row * rowLen;
  float ss = 0.f;
  for (int i = lane; i < rowLen; i += 32) { float v = src[i]; ss += v * v; }
  for (int m = 16; m > 0; m >>= 1) ss += __shfl_xor(ss, m);
  float rinv = rsqrtf(ss);
  for (int i = lane; i < rowLen; i += 32) {
    float v = src[i] * rinv;
    if (dh) dh[(size_t)row * rowLen + i] = (_Float16)v;
    else    df[(size_t)row * rowLen + i] = v;
  }
}

// ---------------------------------------------------------------------------
// Kernel 2: LayerNorm1 + pw conv1 (64 -> 128), WMMA. 128-pixel tile/block.
// ---------------------------------------------------------------------------
__global__ void __launch_bounds__(256)
k_ln1_pw1(const float* __restrict__ inp, const _Float16* __restrict__ wn1,
          const float* __restrict__ b1, const float* __restrict__ ln1w,
          const float* __restrict__ ln1b, _Float16* __restrict__ u) {
  __shared__ alignas(16) float    s_in[NC][128];     // 32 KB
  __shared__ alignas(16) _Float16 s_act[128][72];    // 18.4 KB, 16B-aligned rows
  int b = blockIdx.x >> 9;
  int pixbase = (blockIdx.x & 511) << 7;
  int tid = threadIdx.x;

  for (int idx = tid; idx < NC * 128; idx += 256) {
    int c = idx >> 7, p = idx & 127;
    s_in[c][p] = inp[((size_t)(b * NC + c)) * NHW + pixbase + p];
  }
  __syncthreads();

  if (tid < 128) {
    int p = tid;
    float mu = 0.f;
    for (int c = 0; c < NC; ++c) mu += s_in[c][p];
    mu *= (1.f / 64.f);
    float var = 0.f;
    for (int c = 0; c < NC; ++c) { float d = s_in[c][p] - mu; var += d * d; }
    var *= (1.f / 64.f);
    float rinv = rsqrtf(var + 1e-6f);
    for (int c = 0; c < NC; ++c)
      s_act[p][c] = (_Float16)((s_in[c][p] - mu) * rinv * ln1w[c] + ln1b[c]);
  }
  __syncthreads();

  int wave = tid >> 5, lane = tid & 31;
  int col = lane & 15, mbase = (lane >> 4) * 8, mrow = col;
  int pix = wave * 16 + col;
  const _Float16* actrow = &s_act[pix][0];
  half16 bf0 = load_b_frag(actrow, lane, 0);
  half16 bf1 = load_b_frag(actrow, lane, 1);

#pragma unroll
  for (int mt = 0; mt < 8; ++mt) {
    const _Float16* wrow = wn1 + ((size_t)(b * NDW + mt * 16 + mrow)) * 64;
    float8 acc = bias_acc(b1 + b * NDW + mt * 16, mbase);
    acc = wmma32(load_a_frag(wrow, lane, 0), bf0, acc);
    acc = wmma32(load_a_frag(wrow, lane, 1), bf1, acc);
#pragma unroll
    for (int r = 0; r < 8; ++r) {
      int ch = mt * 16 + mbase + r;
      u[((size_t)(b * NDW + ch)) * NHW + pixbase + pix] = (_Float16)acc[r];
    }
  }
}

// ---------------------------------------------------------------------------
// Kernel 3: depthwise 3x3 (SAME zero-pad) on channel pairs + SimpleGate +
// deterministic per-tile pool partials. 128 threads = 128 pixels (half row).
// ---------------------------------------------------------------------------
__global__ void __launch_bounds__(128)
k_dw_gate_pool(const _Float16* __restrict__ u, const float* __restrict__ wn2,
               const float* __restrict__ b2, _Float16* __restrict__ t,
               float* __restrict__ partial) {
  __shared__ float s_red[4];
  int b = blockIdx.x >> 9;
  int pixbase = (blockIdx.x & 511) << 7;
  int tid = threadIdx.x;
  int wave = tid >> 5, lane = tid & 31;
  int y = pixbase >> 8;
  int gx = (pixbase & 255) + tid;

  for (int c = 0; c < NC; ++c) {
    float a1 = b2[b * NDW + c];
    float a2 = b2[b * NDW + 64 + c];
    const float* w2a = wn2 + ((size_t)(b * NDW + c)) * 9;
    const float* w2b = wn2 + ((size_t)(b * NDW + 64 + c)) * 9;
    const _Float16* ua = u + ((size_t)(b * NDW + c)) * NHW;
    const _Float16* ub = u + ((size_t)(b * NDW + 64 + c)) * NHW;
#pragma unroll
    for (int dy = 0; dy < 3; ++dy) {
      int yy = y + dy - 1;
      if (yy < 0 || yy > 255) continue;
#pragma unroll
      for (int dx = 0; dx < 3; ++dx) {
        int xx = gx + dx - 1;
        if (xx < 0 || xx > 255) continue;
        size_t off = (size_t)yy * 256 + xx;
        a1 += w2a[dy * 3 + dx] * (float)ua[off];
        a2 += w2b[dy * 3 + dx] * (float)ub[off];
      }
    }
    float tv = a1 * a2;  // SimpleGate: first-half * second-half channels
    t[((size_t)(b * NC + c)) * NHW + pixbase + tid] = (_Float16)tv;

    float v = tv;
    for (int m = 16; m > 0; m >>= 1) v += __shfl_xor(v, m);
    if (lane == 0) s_red[wave] = v;
    __syncthreads();
    if (tid == 0)
      partial[(size_t)blockIdx.x * NC + c] = s_red[0] + s_red[1] + s_red[2] + s_red[3];
    __syncthreads();
  }
}

// ---------------------------------------------------------------------------
// Kernel 4: reduce pool partials, compute sca = pooled @ sca_w^T + sca_b.
// ---------------------------------------------------------------------------
__global__ void k_sca(const float* __restrict__ partial,
                      const float* __restrict__ scaw,
                      const float* __restrict__ scab,
                      float* __restrict__ sca) {
  __shared__ float s_p[NC];
  int b = blockIdx.x, o = threadIdx.x;
  float s = 0.f;
  for (int tile = 0; tile < TILES_PER_B; ++tile)
    s += partial[((size_t)(b * TILES_PER_B + tile)) * NC + o];
  s_p[o] = s * (1.0f / (float)NHW);
  __syncthreads();
  float acc = scab[o];
  for (int c = 0; c < NC; ++c) acc += s_p[c] * scaw[o * NC + c];
  sca[b * NC + o] = acc;
}

// ---------------------------------------------------------------------------
// Kernel 5: fused tail: x=t*sca -> pw3 -> y=inp+beta*x -> LN2 -> pw4 ->
// gate -> pw5 -> out = y + gamma*x.  Three WMMA matmuls per 128-px tile.
// ---------------------------------------------------------------------------
__global__ void __launch_bounds__(256)
k_tail(const float* __restrict__ inp, const _Float16* __restrict__ t,
       const float* __restrict__ sca,
       const _Float16* __restrict__ wn3, const float* __restrict__ b3,
       const _Float16* __restrict__ wn4, const float* __restrict__ b4,
       const _Float16* __restrict__ wn5, const float* __restrict__ b5,
       const float* __restrict__ ln2w, const float* __restrict__ ln2b,
       const float* __restrict__ beta, const float* __restrict__ gamma,
       float* __restrict__ out) {
  __shared__ alignas(16) float    s_y[NC][128];      // inp tile, then y
  __shared__ alignas(16) _Float16 s_act[128][72];    // act staging (reused 3x)
  int b = blockIdx.x >> 9;
  int pixbase = (blockIdx.x & 511) << 7;
  int tid = threadIdx.x;

  // P1: load scaled-t (f16 acts) and inp tile.
  for (int idx = tid; idx < NC * 128; idx += 256) {
    int c = idx >> 7, p = idx & 127;
    size_t g = ((size_t)(b * NC + c)) * NHW + pixbase + p;
    s_act[p][c] = (_Float16)((float)t[g] * sca[b * NC + c]);
    s_y[c][p] = inp[g];
  }
  __syncthreads();

  int wave = tid >> 5, lane = tid & 31;
  int col = lane & 15, mbase = (lane >> 4) * 8, mrow = col;
  int pix = wave * 16 + col;
  const _Float16* actrow = &s_act[pix][0];

  // P2: pw3 (64 -> 64)
  half16 bf0 = load_b_frag(actrow, lane, 0);
  half16 bf1 = load_b_frag(actrow, lane, 1);
  float8 acc3[4];
#pragma unroll
  for (int mt = 0; mt < 4; ++mt) {
    const _Float16* wrow = wn3 + ((size_t)(b * NC + mt * 16 + mrow)) * 64;
    float8 acc = bias_acc(b3 + b * NC + mt * 16, mbase);
    acc = wmma32(load_a_frag(wrow, lane, 0), bf0, acc);
    acc = wmma32(load_a_frag(wrow, lane, 1), bf1, acc);
    acc3[mt] = acc;
  }
  // P3: y = inp + beta * x3 (each (c,pix) cell owned by exactly one lane)
#pragma unroll
  for (int mt = 0; mt < 4; ++mt)
#pragma unroll
    for (int r = 0; r < 8; ++r) {
      int c = mt * 16 + mbase + r;
      s_y[c][pix] += beta[c] * acc3[mt][r];
    }
  __syncthreads();

  // P4: LayerNorm2 -> f16 acts
  if (tid < 128) {
    int p = tid;
    float mu = 0.f;
    for (int c = 0; c < NC; ++c) mu += s_y[c][p];
    mu *= (1.f / 64.f);
    float var = 0.f;
    for (int c = 0; c < NC; ++c) { float d = s_y[c][p] - mu; var += d * d; }
    var *= (1.f / 64.f);
    float rinv = rsqrtf(var + 1e-6f);
    for (int c = 0; c < NC; ++c)
      s_act[p][c] = (_Float16)((s_y[c][p] - mu) * rinv * ln2w[c] + ln2b[c]);
  }
  __syncthreads();

  // P5: pw4 (64 -> 128) + SimpleGate (ch c * ch c+64) back into s_act
  bf0 = load_b_frag(actrow, lane, 0);
  bf1 = load_b_frag(actrow, lane, 1);
  float8 acc4[8];
#pragma unroll
  for (int mt = 0; mt < 8; ++mt) {
    const _Float16* wrow = wn4 + ((size_t)(b * NDW + mt * 16 + mrow)) * 64;
    float8 acc = bias_acc(b4 + b * NDW + mt * 16, mbase);
    acc = wmma32(load_a_frag(wrow, lane, 0), bf0, acc);
    acc = wmma32(load_a_frag(wrow, lane, 1), bf1, acc);
    acc4[mt] = acc;
  }
#pragma unroll
  for (int mt = 0; mt < 4; ++mt)
#pragma unroll
    for (int r = 0; r < 8; ++r) {
      int c = mt * 16 + mbase + r;
      s_act[pix][c] = (_Float16)(acc4[mt][r] * acc4[mt + 4][r]);
    }
  __syncthreads();

  // P6: pw5 (64 -> 64), out = y + gamma * x5
  bf0 = load_b_frag(actrow, lane, 0);
  bf1 = load_b_frag(actrow, lane, 1);
#pragma unroll
  for (int mt = 0; mt < 4; ++mt) {
    const _Float16* wrow = wn5 + ((size_t)(b * NC + mt * 16 + mrow)) * 64;
    float8 acc = bias_acc(b5 + b * NC + mt * 16, mbase);
    acc = wmma32(load_a_frag(wrow, lane, 0), bf0, acc);
    acc = wmma32(load_a_frag(wrow, lane, 1), bf1, acc);
#pragma unroll
    for (int r = 0; r < 8; ++r) {
      int c = mt * 16 + mbase + r;
      out[((size_t)(b * NC + c)) * NHW + pixbase + pix] =
          s_y[c][pix] + gamma[c] * acc[r];
    }
  }
}

// ---------------------------------------------------------------------------
extern "C" void kernel_launch(void* const* d_in, const int* in_sizes, int n_in,
                              void* d_out, int out_size, void* d_ws, size_t ws_size,
                              hipStream_t stream) {
  (void)in_sizes; (void)n_in; (void)out_size; (void)ws_size;
  const float* inp  = (const float*)d_in[0];
  const float* w1   = (const float*)d_in[1];
  const float* b1   = (const float*)d_in[2];
  const float* w2   = (const float*)d_in[3];
  const float* b2   = (const float*)d_in[4];
  const float* w3   = (const float*)d_in[5];
  const float* b3   = (const float*)d_in[6];
  const float* w4   = (const float*)d_in[7];
  const float* b4   = (const float*)d_in[8];
  const float* w5   = (const float*)d_in[9];
  const float* b5   = (const float*)d_in[10];
  const float* ln1w = (const float*)d_in[11];
  const float* ln1b = (const float*)d_in[12];
  const float* ln2w = (const float*)d_in[13];
  const float* ln2b = (const float*)d_in[14];
  const float* scaw = (const float*)d_in[15];
  const float* scab = (const float*)d_in[16];
  const float* beta = (const float*)d_in[17];
  const float* gamma= (const float*)d_in[18];
  float* out = (float*)d_out;

  // Workspace layout (assumes ws_size >= ~203 MB; all regions fully written
  // before being read, no cross-call state).
  char* ws = (char*)d_ws;
  size_t off = 0;
  _Float16* wn1 = (_Float16*)(ws + off); off += (size_t)NB * NDW * 64 * 2;  // 128K
  _Float16* wn3 = (_Float16*)(ws + off); off += (size_t)NB * NC  * 64 * 2;  // 64K
  _Float16* wn4 = (_Float16*)(ws + off); off += (size_t)NB * NDW * 64 * 2;  // 128K
  _Float16* wn5 = (_Float16*)(ws + off); off += (size_t)NB * NC  * 64 * 2;  // 64K
  float*    wn2 = (float*)   (ws + off); off += (size_t)NB * NDW * 9 * 4;   // 36K
  float*    part= (float*)   (ws + off); off += (size_t)NB * TILES_PER_B * NC * 4; // 1M
  float*    sca = (float*)   (ws + off); off += (size_t)NB * NC * 4;        // 2K
  off = (off + 255) & ~(size_t)255;
  _Float16* u   = (_Float16*)(ws + off); off += (size_t)NB * NDW * NHW * 2; // 128M
  _Float16* t   = (_Float16*)(ws + off); off += (size_t)NB * NC  * NHW * 2; // 64M

  // 1) normalize weights
  k_prep<<<NB * NDW, 32, 0, stream>>>(w1, wn1, nullptr, 64);
  k_prep<<<NB * NDW, 32, 0, stream>>>(w2, nullptr, wn2, 9);
  k_prep<<<NB * NC , 32, 0, stream>>>(w3, wn3, nullptr, 64);
  k_prep<<<NB * NDW, 32, 0, stream>>>(w4, wn4, nullptr, 64);
  k_prep<<<NB * NC , 32, 0, stream>>>(w5, wn5, nullptr, 64);

  dim3 grid(NB * TILES_PER_B);
  // 2) LN1 + pw1 (WMMA)
  k_ln1_pw1<<<grid, 256, 0, stream>>>(inp, wn1, b1, ln1w, ln1b, u);
  // 3) depthwise 3x3 + gate + pool partials
  k_dw_gate_pool<<<grid, 128, 0, stream>>>(u, wn2, b2, t, part);
  // 4) SCA
  k_sca<<<NB, 64, 0, stream>>>(part, scaw, scab, sca);
  // 5) fused tail (3 WMMA matmuls)
  k_tail<<<grid, 256, 0, stream>>>(inp, t, sca, wn3, b3, wn4, b4, wn5, b5,
                                   ln2w, ln2b, beta, gamma, out);
}
